// SlidingWindowAttention_85409719648568
// MI455X (gfx1250) — matrix operand order repaired and checked by reading, not compiled, not run on hardware
//
#include <hip/hip_runtime.h>
#include <hip/hip_bf16.h>

// ---------------------------------------------------------------------------
// Sliding-window attention (NUM_HEADS=16, HEAD_DIM=128, WINDOW=512, S=4096)
// bf16 WMMA pipeline for gfx1250 (CDNA5, wave32):
//   x,W (f32) -> bf16; Q/K/V projections via v_wmma_f32_16x16x32_bf16 with
//   async global->LDS staging (ASYNCcnt) and double-buffered LDS tiles;
//   RoPE on Q,K; flash attention with WMMA QK^T and PV; output proj -> f32.
// ---------------------------------------------------------------------------

typedef __attribute__((ext_vector_type(16))) __bf16 v16bf;
typedef __attribute__((ext_vector_type(8)))  float  v8f;

#define S_LEN 4096
#define DIM   2048
#define NH    16
#define DH    128
#define WIN   512

__device__ __forceinline__ unsigned short f2bfbits(float f) {
  unsigned u = __builtin_bit_cast(unsigned, f);
  u += 0x7FFFu + ((u >> 16) & 1u);           // round-to-nearest-even
  return (unsigned short)(u >> 16);
}
__device__ __forceinline__ float bfbits2f(unsigned short h) {
  unsigned u = ((unsigned)h) << 16;
  return __builtin_bit_cast(float, u);
}

union FragA { v16bf v; int4 q[2]; };

// --- gfx1250 async global->LDS copy (ASYNCcnt-tracked, no VGPR staging) ----
// Low 32 bits of a generic pointer to __shared__ = LDS byte offset (aperture
// layout: isLDS when high bits match SHARED_BASE).
__device__ __forceinline__ unsigned lds_off(const void* p) {
  return (unsigned)(size_t)p;
}
// GVS addressing: mem = SADDR(u64) + VADDR(i32) ; 16B per lane direct to LDS.
__device__ __forceinline__ void async_ld_b128(unsigned ldsAddr,
                                              const unsigned short* base,
                                              unsigned byteOff) {
  asm volatile("global_load_async_to_lds_b128 %0, %1, %2"
               :: "v"(ldsAddr), "v"(byteOff), "s"(base)
               : "memory");
}
__device__ __forceinline__ void wait_async() {
  asm volatile("s_wait_asynccnt 0x0" ::: "memory");
}

// ---------------------------------------------------------------------------
// f32 -> bf16 conversion (vectorized, n divisible by 4)
// ---------------------------------------------------------------------------
__global__ __launch_bounds__(256)
void cvt_f32_bf16(const float* __restrict__ src, unsigned short* __restrict__ dst, int n)
{
  int i = (blockIdx.x * 256 + threadIdx.x) * 4;
  if (i + 3 < n) {
    float4 f = *reinterpret_cast<const float4*>(src + i);
    ushort4 o;
    o.x = f2bfbits(f.x); o.y = f2bfbits(f.y);
    o.z = f2bfbits(f.z); o.w = f2bfbits(f.w);
    *reinterpret_cast<ushort4*>(dst + i) = o;
  }
}

// ---------------------------------------------------------------------------
// GEMM  C[M,N] = A[M,K] * W[N,K]^T   (A,W bf16; accumulate f32)
// MODE 0: store bf16 row-major   (Q,K projections)
// MODE 1: store bf16 transposed C[n*M+m]  (V projection -> V^T for PV WMMA)
// MODE 2: store f32 row-major    (output projection -> d_out)
// Block: 256 thr = 8 waves; tile 128x128; K staged 64-deep, double-buffered
// LDS fed by global_load_async_to_lds_b128 (tile i+1 streams while tile i
// feeds the WMMA pipe). Per wave: 32(M) x 64(N), 16 WMMAs per K-stage.
// ---------------------------------------------------------------------------
template<int MODE>
__global__ __launch_bounds__(256)
void gemm_bf16_wmma(const unsigned short* __restrict__ A,
                    const unsigned short* __restrict__ W,
                    void* __restrict__ Cout,
                    int M, int N, int K)
{
  constexpr int LT = 72;  // LDS row stride (bf16 elems), 144B = 9x16B, conflict-skewed
  __shared__ __align__(16) unsigned short As[2][128 * LT];
  __shared__ __align__(16) unsigned short Bs[2][128 * LT];

  const int tid  = threadIdx.x;
  const int wid  = tid >> 5;
  const int lane = tid & 31;
  const int half = lane >> 4;   // lanes 16-31 hold the other K/M half per ISA layout
  const int lr   = lane & 15;
  const int wm   = wid >> 1;    // 0..3 : wave M position
  const int wn   = wid & 1;     // 0..1 : wave N position
  const int m0 = blockIdx.y * 128;
  const int n0 = blockIdx.x * 128;

  const v8f vzero = {0.f,0.f,0.f,0.f,0.f,0.f,0.f,0.f};
  v8f acc[2][4];
#pragma unroll
  for (int mt = 0; mt < 2; ++mt)
#pragma unroll
    for (int nt = 0; nt < 4; ++nt) acc[mt][nt] = vzero;

  // issue one 128x64 K-stage (A and W tiles) into LDS buffer `buf`
  auto issue = [&](int buf, int k0) {
#pragma unroll
    for (int i = 0; i < 4; ++i) {                 // 1024 int4 per tile / 256 thr
      int c = tid + 256 * i;
      int row = c >> 3, seg = c & 7;
      async_ld_b128(lds_off(&As[buf][row * LT + seg * 8]), A,
                    (unsigned)((((size_t)(m0 + row)) * K + k0 + seg * 8) * 2));
      async_ld_b128(lds_off(&Bs[buf][row * LT + seg * 8]), W,
                    (unsigned)((((size_t)(n0 + row)) * K + k0 + seg * 8) * 2));
    }
  };

  issue(0, 0);
  wait_async();
  __syncthreads();

  const int NK = K >> 6;
  for (int ki = 0; ki < NK; ++ki) {
    const int cur = ki & 1;
    if (ki + 1 < NK) issue(cur ^ 1, (ki + 1) << 6);   // overlap next stage

    const unsigned short* Ab = As[cur];
    const unsigned short* Bb = Bs[cur];
#pragma unroll
    for (int ks = 0; ks < 2; ++ks) {                  // two K=32 WMMA steps
      FragA af[2];
#pragma unroll
      for (int mt = 0; mt < 2; ++mt) {
        // A 16x32 bf16 layout: lane holds row (lane&15), K = base..base+7, base+16..+23
        int ab = (wm * 32 + mt * 16 + lr) * LT + ks * 32 + half * 8;
        af[mt].q[0] = *reinterpret_cast<const int4*>(&Ab[ab]);
        af[mt].q[1] = *reinterpret_cast<const int4*>(&Ab[ab + 16]);
      }
#pragma unroll
      for (int nt = 0; nt < 4; ++nt) {
        // B 32x16: lane holds column n=(lane&15), K = base..base+7, base+16..+23 (base=half*8)
        FragA bfr;
        int bb = (wn * 64 + nt * 16 + lr) * LT + ks * 32 + half * 8;
        bfr.q[0] = *reinterpret_cast<const int4*>(&Bb[bb]);
        bfr.q[1] = *reinterpret_cast<const int4*>(&Bb[bb + 16]);
#pragma unroll
        for (int mt = 0; mt < 2; ++mt) {
          acc[mt][nt] = __builtin_amdgcn_wmma_f32_16x16x32_bf16(
              false, af[mt].v, false, bfr.v, (short)0, acc[mt][nt], false, false);
        }
      }
    }
    wait_async();      // own wave's async writes to next buffer complete
    __syncthreads();   // all waves: next tile visible, current tile consumed
  }

#pragma unroll
  for (int mt = 0; mt < 2; ++mt)
#pragma unroll
    for (int nt = 0; nt < 4; ++nt)
#pragma unroll
      for (int r = 0; r < 8; ++r) {
        int row = m0 + wm * 32 + mt * 16 + r + half * 8;  // C layout: M=r(+8 upper half)
        int col = n0 + wn * 64 + nt * 16 + lr;            // N = lane&15
        float v = acc[mt][nt][r];
        if (MODE == 0)
          ((unsigned short*)Cout)[(size_t)row * N + col] = f2bfbits(v);
        else if (MODE == 1)
          ((unsigned short*)Cout)[(size_t)col * M + row] = f2bfbits(v);
        else
          ((float*)Cout)[(size_t)row * N + col] = v;
      }
}

// ---------------------------------------------------------------------------
// RoPE on Q and K (interleaved pairs, base 10000), in-place on bf16 buffers.
// grid (S, H), block 64 (one pair per thread).
// ---------------------------------------------------------------------------
__global__ __launch_bounds__(64)
void rope_bf16(unsigned short* __restrict__ q, unsigned short* __restrict__ k)
{
  const int s = blockIdx.x;
  const int h = blockIdx.y;
  const int i = threadIdx.x;                       // pair index 0..63
  const float LN_BASE = 9.2103403719761836f;       // ln(10000)
  float inv_freq = __expf(-((float)(2 * i) * (LN_BASE / (float)DH)));
  float th = (float)s * inv_freq;
  float c = cosf(th), sn = sinf(th);
  size_t base = (size_t)s * DIM + (size_t)h * DH + 2 * i;
  float q1 = bfbits2f(q[base]), q2 = bfbits2f(q[base + 1]);
  q[base]     = f2bfbits(q1 * c - q2 * sn);
  q[base + 1] = f2bfbits(q1 * sn + q2 * c);
  float k1 = bfbits2f(k[base]), k2 = bfbits2f(k[base + 1]);
  k[base]     = f2bfbits(k1 * c - k2 * sn);
  k[base + 1] = f2bfbits(k1 * sn + k2 * c);
}

// ---------------------------------------------------------------------------
// Flash sliding-window attention. grid (S/64, H), block 128 (4 waves).
// Each wave owns 16 queries x full 128-d output (8 f32 acc tiles).
// K/V tiles streamed to LDS with async copies; per 64-key block:
// 16 WMMAs (QK^T) + online softmax + 16 WMMAs (PV).
// V is consumed transposed ([feature][seq]) so B-frags are contiguous.
// ---------------------------------------------------------------------------
__global__ __launch_bounds__(128)
void swa_attn_wmma(const unsigned short* __restrict__ Qg,
                   const unsigned short* __restrict__ Kg,
                   const unsigned short* __restrict__ Vt,
                   unsigned short* __restrict__ Og)
{
  __shared__ __align__(16) unsigned short Ks[64 * 136];   // [key][d]
  __shared__ __align__(16) unsigned short Vs[128 * 72];   // [d][key]
  __shared__ __align__(16) unsigned short Ps[4 * 16 * 72];// per-wave P relayout

  const int h    = blockIdx.y;
  const int Q0   = blockIdx.x * 64;
  const int tid  = threadIdx.x;
  const int wid  = tid >> 5;
  const int lane = tid & 31;
  const int half = lane >> 4;
  const int lr   = lane & 15;
  const int qrow = Q0 + wid * 16 + lr;

  // Q A-frags: 4 d-tiles of 16x32, straight from global (contiguous 16B loads)
  FragA qf[4];
#pragma unroll
  for (int dt = 0; dt < 4; ++dt) {
    const unsigned short* p = &Qg[(size_t)qrow * DIM + h * DH + dt * 32 + half * 8];
    qf[dt].q[0] = *reinterpret_cast<const int4*>(p);
    qf[dt].q[1] = *reinterpret_cast<const int4*>(p + 16);
  }

  const v8f vzero = {0.f,0.f,0.f,0.f,0.f,0.f,0.f,0.f};
  v8f o[8];
  float m_run[8], l_run[8], alpha[8];
#pragma unroll
  for (int dt = 0; dt < 8; ++dt) o[dt] = vzero;
#pragma unroll
  for (int r = 0; r < 8; ++r) { m_run[r] = -3.0e38f; l_run[r] = 0.f; }

  const int kbs = (Q0 > (WIN - 1)) ? ((Q0 - (WIN - 1)) & ~63) : 0;
  for (int kb = kbs; kb <= Q0; kb += 64) {
    __syncthreads();                       // previous tile fully consumed
#pragma unroll
    for (int i = 0; i < 8; ++i) {
      int c = tid + 128 * i;
      { int row = c >> 4, seg = c & 15;    // K tile: 64 keys x 128 d
        async_ld_b128(lds_off(&Ks[row * 136 + seg * 8]), Kg,
                      (unsigned)((((size_t)(kb + row)) * DIM + h * DH + seg * 8) * 2)); }
      { int row = c >> 3, seg = c & 7;     // V^T tile: 128 d x 64 keys
        async_ld_b128(lds_off(&Vs[row * 72 + seg * 8]), Vt,
                      (unsigned)((((size_t)(h * DH + row)) * S_LEN + kb + seg * 8) * 2)); }
    }
    wait_async();
    __syncthreads();

    // ---- scores S = Q K^T (4 n-tiles of 16 keys, K=128 over d) ----
    v8f sa[4];
#pragma unroll
    for (int t = 0; t < 4; ++t) sa[t] = vzero;
#pragma unroll
    for (int t = 0; t < 4; ++t)
#pragma unroll
      for (int dt = 0; dt < 4; ++dt) {
        FragA kf;
        int bb = (t * 16 + lr) * 136 + dt * 32 + half * 8;
        kf.q[0] = *reinterpret_cast<const int4*>(&Ks[bb]);
        kf.q[1] = *reinterpret_cast<const int4*>(&Ks[bb + 16]);
        sa[t] = __builtin_amdgcn_wmma_f32_16x16x32_bf16(
            false, qf[dt].v, false, kf.v, (short)0, sa[t], false, false);
      }

    // ---- mask + online softmax (rows live in 16-lane halves; shfl_xor reduce) ----
    const float scale = 0.088388347648318447f;   // 1/sqrt(128)
#pragma unroll
    for (int r = 0; r < 8; ++r) {
      const int qi = Q0 + wid * 16 + r + half * 8;
      float rm = -3.0e38f;
#pragma unroll
      for (int t = 0; t < 4; ++t) {
        int ki = kb + t * 16 + lr;
        float s = sa[t][r] * scale;
        bool valid = (ki <= qi) && (qi - ki < WIN);
        s = valid ? s : -3.0e38f;
        sa[t][r] = s;
        rm = fmaxf(rm, s);
      }
#pragma unroll
      for (int off = 1; off < 16; off <<= 1)
        rm = fmaxf(rm, __shfl_xor(rm, off, 32));
      float mn = fmaxf(m_run[r], rm);
      float a  = __expf(m_run[r] - mn);
      float rs = 0.f;
#pragma unroll
      for (int t = 0; t < 4; ++t) {
        float s = sa[t][r];
        float p = (s < -1.0e37f) ? 0.f : __expf(s - mn);
        sa[t][r] = p;
        rs += p;
      }
#pragma unroll
      for (int off = 1; off < 16; off <<= 1)
        rs += __shfl_xor(rs, off, 32);
      l_run[r] = l_run[r] * a + rs;
      m_run[r] = mn;
      alpha[r] = a;
    }
#pragma unroll
    for (int dt = 0; dt < 8; ++dt)
#pragma unroll
      for (int r = 0; r < 8; ++r) o[dt][r] *= alpha[r];

    // ---- relayout P (C-layout -> A-layout) through per-wave LDS ----
    unsigned short* pl = &Ps[wid * 16 * 72];
#pragma unroll
    for (int t = 0; t < 4; ++t)
#pragma unroll
      for (int r = 0; r < 8; ++r)
        pl[(r + half * 8) * 72 + t * 16 + lr] = f2bfbits(sa[t][r]);

    // ---- O += P V  (2 K-steps of 32 keys x 8 d-tiles) ----
#pragma unroll
    for (int kt = 0; kt < 2; ++kt) {
      FragA pf;
      int pb = lr * 72 + kt * 32 + half * 8;
      pf.q[0] = *reinterpret_cast<const int4*>(&pl[pb]);
      pf.q[1] = *reinterpret_cast<const int4*>(&pl[pb + 16]);
#pragma unroll
      for (int dt = 0; dt < 8; ++dt) {
        FragA vf;
        int vb = (dt * 16 + lr) * 72 + kt * 32 + half * 8;
        vf.q[0] = *reinterpret_cast<const int4*>(&Vs[vb]);
        vf.q[1] = *reinterpret_cast<const int4*>(&Vs[vb + 16]);
        o[dt] = __builtin_amdgcn_wmma_f32_16x16x32_bf16(
            false, pf.v, false, vf.v, (short)0, o[dt], false, false);
      }
    }
  }

  // ---- normalize and store bf16 attention output ----
#pragma unroll
  for (int r = 0; r < 8; ++r)
    alpha[r] = (l_run[r] > 0.f) ? (1.f / l_run[r]) : 0.f;
#pragma unroll
  for (int dt = 0; dt < 8; ++dt)
#pragma unroll
    for (int r = 0; r < 8; ++r) {
      int qi = Q0 + wid * 16 + r + half * 8;
      Og[(size_t)qi * DIM + h * DH + dt * 16 + lr] = f2bfbits(o[dt][r] * alpha[r]);
    }
}

// ---------------------------------------------------------------------------
// Host orchestration
// ---------------------------------------------------------------------------
extern "C" void kernel_launch(void* const* d_in, const int* in_sizes, int n_in,
                              void* d_out, int out_size, void* d_ws, size_t ws_size,
                              hipStream_t stream)
{
  const float* x  = (const float*)d_in[0];
  const float* Wq = (const float*)d_in[1];
  const float* Wk = (const float*)d_in[2];
  const float* Wv = (const float*)d_in[3];
  const float* Wo = (const float*)d_in[4];

  const size_t XE = (size_t)S_LEN * DIM;   // 8M elems
  const size_t WE = (size_t)DIM * DIM;     // 4M elems
  const size_t need = (5 * XE + 4 * WE) * sizeof(unsigned short);  // ~112 MiB
  if (ws_size < need) return;

  unsigned short* xb  = (unsigned short*)d_ws;
  unsigned short* wqb = xb  + XE;
  unsigned short* wkb = wqb + WE;
  unsigned short* wvb = wkb + WE;
  unsigned short* wob = wvb + WE;
  unsigned short* qb  = wob + WE;
  unsigned short* kbf = qb  + XE;
  unsigned short* vtb = kbf + XE;          // V stored transposed [feature][seq]
  unsigned short* ab  = vtb + XE;

  // f32 -> bf16
  cvt_f32_bf16<<<dim3((unsigned)(XE / 1024)), 256, 0, stream>>>(x,  xb,  (int)XE);
  cvt_f32_bf16<<<dim3((unsigned)(WE / 1024)), 256, 0, stream>>>(Wq, wqb, (int)WE);
  cvt_f32_bf16<<<dim3((unsigned)(WE / 1024)), 256, 0, stream>>>(Wk, wkb, (int)WE);
  cvt_f32_bf16<<<dim3((unsigned)(WE / 1024)), 256, 0, stream>>>(Wv, wvb, (int)WE);
  cvt_f32_bf16<<<dim3((unsigned)(WE / 1024)), 256, 0, stream>>>(Wo, wob, (int)WE);

  // Q/K/V projections (V written transposed)
  dim3 gg(DIM / 128, S_LEN / 128);  // (16, 32)
  gemm_bf16_wmma<0><<<gg, 256, 0, stream>>>(xb, wqb, qb,  S_LEN, DIM, DIM);
  gemm_bf16_wmma<0><<<gg, 256, 0, stream>>>(xb, wkb, kbf, S_LEN, DIM, DIM);
  gemm_bf16_wmma<1><<<gg, 256, 0, stream>>>(xb, wvb, vtb, S_LEN, DIM, DIM);

  // RoPE on Q,K
  rope_bf16<<<dim3(S_LEN, NH), 64, 0, stream>>>(qb, kbf);

  // Flash sliding-window attention
  swa_attn_wmma<<<dim3(S_LEN / 64, NH), 128, 0, stream>>>(qb, kbf, vtb, ab);

  // Output projection -> f32 d_out
  gemm_bf16_wmma<2><<<gg, 256, 0, stream>>>(ab, wob, d_out, S_LEN, DIM, DIM);
}
